// GCNPolicy_63874753626724
// MI455X (gfx1250) — matrix operand order repaired and checked by reading, not compiled, not run on hardware
//
#include <hip/hip_runtime.h>

#define SCALE_F 0.4251202479144762f

typedef __attribute__((ext_vector_type(16))) _Float16 v16h;
typedef __attribute__((ext_vector_type(8)))  float    v8f;

__device__ __forceinline__ void atomAddF(float* p, float v) {
    __hip_atomic_fetch_add(p, v, __ATOMIC_RELAXED, __HIP_MEMORY_SCOPE_AGENT);
}

__device__ __forceinline__ float reluf(float x) { return x > 0.0f ? x : 0.0f; }

// Pack the ISA 16-bit A-fragment (16x32 K-block) for one lane:
// VGPR0-3 -> K = half*8 + 0..7 ; VGPR4-7 -> K = 16 + half*8 + 0..7
__device__ __forceinline__ v16h load_a_frag(const float* __restrict__ arow) {
    const float4* p0 = (const float4*)(arow);        // K offsets 0..7
    const float4* p1 = (const float4*)(arow + 16);   // K offsets 16..23
    float4 a0 = p0[0], a1 = p0[1], a2 = p1[0], a3 = p1[1];
    v16h a;
    a[0]=(_Float16)a0.x; a[1]=(_Float16)a0.y; a[2]=(_Float16)a0.z; a[3]=(_Float16)a0.w;
    a[4]=(_Float16)a1.x; a[5]=(_Float16)a1.y; a[6]=(_Float16)a1.z; a[7]=(_Float16)a1.w;
    a[8]=(_Float16)a2.x; a[9]=(_Float16)a2.y; a[10]=(_Float16)a2.z; a[11]=(_Float16)a2.w;
    a[12]=(_Float16)a3.x; a[13]=(_Float16)a3.y; a[14]=(_Float16)a3.z; a[15]=(_Float16)a3.w;
    return a;
}

// ---------------- utility: zero a float region (float4) ----------------
__global__ void k_zero4(float4* __restrict__ p, long n4) {
    long i = (long)blockIdx.x * blockDim.x + threadIdx.x;
    long stride = (long)gridDim.x * blockDim.x;
    float4 z = make_float4(0.f, 0.f, 0.f, 0.f);
    for (; i < n4; i += stride) p[i] = z;
}

// ---------------- edge-weight norm: sum of squares ----------------
__global__ void k_norm_sq(const float* __restrict__ ew, float* __restrict__ sumsq, int E) {
    __shared__ float sm[8];
    float s = 0.0f;
    for (int i = blockIdx.x * blockDim.x + threadIdx.x; i < E;
         i += gridDim.x * blockDim.x) {
        float v = ew[i];
        s += v * v;
    }
    #pragma unroll
    for (int off = 16; off > 0; off >>= 1) s += __shfl_down(s, off, 32);
    int lane = threadIdx.x & 31, wv = threadIdx.x >> 5;
    if (lane == 0) sm[wv] = s;
    __syncthreads();
    if (threadIdx.x == 0) {
        float tot = 0.0f;
        int nw = blockDim.x >> 5;
        for (int w = 0; w < nw; ++w) tot += sm[w];
        atomAddF(sumsq, tot);
    }
}

__global__ void k_norm_finalize(float* __restrict__ scal) {
    if (threadIdx.x == 0 && blockIdx.x == 0) scal[1] = rsqrtf(scal[0]);
}

// ---------------- dense GEMM + bias + relu via WMMA ----------------
// out[nrows x 64] = relu(A[nrows x K] @ W[K x 64] + bias[64]); K in {32,64}
// One wave computes a full 16x64 row stripe (4 col tiles, A fragment reused 4x).
// W^T staged in LDS as f16 so each B fragment is two contiguous 16B ds reads.
__global__ __launch_bounds__(256)
void k_gemm_bias_relu(const float* __restrict__ A, const float* __restrict__ W,
                      const float* __restrict__ bias, float* __restrict__ out,
                      int nrows, int K) {
    __shared__ __align__(32) _Float16 sWT[64 * 80];   // max K=64, stride K+16
    const int stride = K + 16;                        // halves; keeps 32B alignment
    for (int i = threadIdx.x; i < K * 64; i += blockDim.x) {
        int k = i >> 6, n = i & 63;                   // W row-major [k][n]
        sWT[n * stride + k] = (_Float16)W[i];
    }
    __syncthreads();

    const int lane = threadIdx.x & 31;
    const int wave = threadIdx.x >> 5;
    const int rowTile = blockIdx.x * (blockDim.x >> 5) + wave;
    if (rowTile >= (nrows >> 4)) return;              // wave-uniform exit
    const int m    = lane & 15;
    const int half = lane >> 4;

    v8f acc[4] = {v8f{}, v8f{}, v8f{}, v8f{}};
    const float* arowBase = A + (size_t)(rowTile * 16 + m) * K + half * 8;
    for (int kb = 0; kb < K; kb += 32) {
        v16h a = load_a_frag(arowBase + kb);
        #pragma unroll
        for (int ct = 0; ct < 4; ++ct) {
            int n = ct * 16 + m;                      // B column for this lane
            v16h bf;
            #pragma unroll
            for (int s = 0; s < 16; ++s) bf[s] = sWT[n * stride + kb + half * 8 + (s & 7) + (s >> 3) * 16];
            acc[ct] = __builtin_amdgcn_wmma_f32_16x16x32_f16(
                false, a, false, bf, (short)0, acc[ct], false, false);
        }
    }
    #pragma unroll
    for (int ct = 0; ct < 4; ++ct) {
        const int n = ct * 16 + m;
        const float bn = bias[n];
        #pragma unroll
        for (int j = 0; j < 8; ++j) {                 // C VGPR j -> row half*8 + j
            int row = rowTile * 16 + half * 8 + j;
            out[(size_t)row * 64 + n] = reluf(acc[ct][j] + bn);
        }
    }
}

// ---------------- fused output head: out[r] = relu(X[r,:]@W1 + b1) . w2 --------
__global__ __launch_bounds__(256)
void k_head(const float* __restrict__ X, const float* __restrict__ W1,
            const float* __restrict__ b1, const float* __restrict__ w2,
            float* __restrict__ out, int nrows) {
    const int K = 64;
    __shared__ __align__(32) _Float16 sWT[64 * 80];
    const int stride = K + 16;
    for (int i = threadIdx.x; i < K * 64; i += blockDim.x) {
        int k = i >> 6, n = i & 63;
        sWT[n * stride + k] = (_Float16)W1[i];
    }
    __syncthreads();

    const int lane = threadIdx.x & 31;
    const int wave = threadIdx.x >> 5;
    const int rowTile = blockIdx.x * (blockDim.x >> 5) + wave;
    if (rowTile >= (nrows >> 4)) return;
    const int m    = lane & 15;
    const int half = lane >> 4;

    v8f acc[4] = {v8f{}, v8f{}, v8f{}, v8f{}};
    const float* arowBase = X + (size_t)(rowTile * 16 + m) * K + half * 8;
    #pragma unroll
    for (int kb = 0; kb < K; kb += 32) {
        v16h a = load_a_frag(arowBase + kb);
        #pragma unroll
        for (int ct = 0; ct < 4; ++ct) {
            int n = ct * 16 + m;
            v16h bf;
            #pragma unroll
            for (int s = 0; s < 16; ++s) bf[s] = sWT[n * stride + kb + half * 8 + (s & 7) + (s >> 3) * 16];
            acc[ct] = __builtin_amdgcn_wmma_f32_16x16x32_f16(
                false, a, false, bf, (short)0, acc[ct], false, false);
        }
    }
    // d[j] = sum_n relu(C[row=half*8+j][n] + b1[n]) * w2[n]
    float d[8] = {0.f, 0.f, 0.f, 0.f, 0.f, 0.f, 0.f, 0.f};
    #pragma unroll
    for (int ct = 0; ct < 4; ++ct) {
        const int n = ct * 16 + m;
        const float bn = b1[n], wn = w2[n];
        #pragma unroll
        for (int j = 0; j < 8; ++j) d[j] += reluf(acc[ct][j] + bn) * wn;
    }
    #pragma unroll
    for (int mask = 1; mask <= 8; mask <<= 1) {       // reduce across 16-lane group
        #pragma unroll
        for (int j = 0; j < 8; ++j) d[j] += __shfl_xor(d[j], mask, 32);
    }
    if ((lane & 15) == 0) {
        #pragma unroll
        for (int j = 0; j < 8; ++j)
            out[rowTile * 16 + half * 8 + j] = d[j];
    }
}

// ---------------- SPMM (scatter-add): out[src] += w * feat[dst] ----------------
// 16 threads per edge, float4 (16B) per thread: 64 contiguous floats per edge.
__global__ void k_spmm1(const int* __restrict__ ei, const float* __restrict__ ew,
                        const float* __restrict__ scal,
                        const float* __restrict__ feat, float* __restrict__ out, int E) {
    long tid = (long)blockIdx.x * blockDim.x + threadIdx.x;
    int e = (int)(tid >> 4);
    if (e >= E) return;
    int g   = (int)(tid & 15);
    if (g == 0) {                                     // speculative prefetch ahead
        __builtin_prefetch(ei + 2 * e + 8192, 0, 0);
        __builtin_prefetch(ew + e + 4096, 0, 0);
    }
    int src = ei[2 * e];
    int dst = ei[2 * e + 1];
    float w = ew[e] * scal[1];
    float4 f = *(const float4*)(feat + (size_t)dst * 64 + g * 4);
    float* o = out + (size_t)src * 64 + g * 4;
    atomAddF(o + 0, w * f.x);
    atomAddF(o + 1, w * f.y);
    atomAddF(o + 2, w * f.z);
    atomAddF(o + 3, w * f.w);
}

// out[src] += w * (2*var - var_k)[dst]
__global__ void k_spmm2(const int* __restrict__ ei, const float* __restrict__ ew,
                        const float* __restrict__ scal,
                        const float* __restrict__ var, const float* __restrict__ var_k,
                        float* __restrict__ out, int E) {
    long tid = (long)blockIdx.x * blockDim.x + threadIdx.x;
    int e = (int)(tid >> 4);
    if (e >= E) return;
    int g   = (int)(tid & 15);
    if (g == 0) {
        __builtin_prefetch(ei + 2 * e + 8192, 0, 0);
        __builtin_prefetch(ew + e + 4096, 0, 0);
    }
    int src = ei[2 * e];
    int dst = ei[2 * e + 1];
    float w = ew[e] * scal[1];
    size_t off = (size_t)dst * 64 + g * 4;
    float4 fv = *(const float4*)(var + off);
    float4 fk = *(const float4*)(var_k + off);
    float* o = out + (size_t)src * 64 + g * 4;
    atomAddF(o + 0, w * (2.0f * fv.x - fk.x));
    atomAddF(o + 1, w * (2.0f * fv.y - fk.y));
    atomAddF(o + 2, w * (2.0f * fv.z - fk.z));
    atomAddF(o + 3, w * (2.0f * fv.w - fk.w));
}

// ---------------- elementwise layer updates (float4, fused conv re-zero) -------
// var_k = var; var = relu((var + t*(c - conv)) * SCALE); conv = 0
__global__ void k_update_var(float4* __restrict__ var, float4* __restrict__ var_k,
                             float4* __restrict__ conv, const float* __restrict__ c,
                             const float* __restrict__ temps, int tidx, long total4) {
    long i = (long)blockIdx.x * blockDim.x + threadIdx.x;
    if (i >= total4) return;
    float t  = temps[tidx];
    float cc = c[i >> 4];                             // 16 float4 per row of 64
    float4 vk = var[i];
    float4 cv = conv[i];
    float4 x;
    x.x = reluf((vk.x + t * (cc - cv.x)) * SCALE_F);
    x.y = reluf((vk.y + t * (cc - cv.y)) * SCALE_F);
    x.z = reluf((vk.z + t * (cc - cv.z)) * SCALE_F);
    x.w = reluf((vk.w + t * (cc - cv.w)) * SCALE_F);
    var_k[i] = vk;
    var[i]   = x;
    conv[i]  = make_float4(0.f, 0.f, 0.f, 0.f);
}

// cons = relu(cons - t*(b - conv2)); conv2 = 0
__global__ void k_update_cons(float4* __restrict__ cons, float4* __restrict__ conv2,
                              const float* __restrict__ b,
                              const float* __restrict__ temps, int tidx, long total4) {
    long i = (long)blockIdx.x * blockDim.x + threadIdx.x;
    if (i >= total4) return;
    float t  = temps[tidx];
    float bb = b[i >> 4];
    float4 cs = cons[i];
    float4 cv = conv2[i];
    float4 x;
    x.x = reluf(cs.x - t * (bb - cv.x));
    x.y = reluf(cs.y - t * (bb - cv.y));
    x.z = reluf(cs.z - t * (bb - cv.z));
    x.w = reluf(cs.w - t * (bb - cv.w));
    cons[i]  = x;
    conv2[i] = make_float4(0.f, 0.f, 0.f, 0.f);
}

extern "C" void kernel_launch(void* const* d_in, const int* in_sizes, int n_in,
                              void* d_out, int out_size, void* d_ws, size_t ws_size,
                              hipStream_t stream) {
    const float* con_feat = (const float*)d_in[0];
    const float* var_feat = (const float*)d_in[1];
    const float* edge_w   = (const float*)d_in[2];
    const float* c        = (const float*)d_in[3];
    const float* b        = (const float*)d_in[4];
    const float* W_cons   = (const float*)d_in[5];
    const float* b_cons   = (const float*)d_in[6];
    const float* W_var    = (const float*)d_in[7];
    const float* b_var    = (const float*)d_in[8];
    const float* t_ctov   = (const float*)d_in[9];
    const float* t_vtoc   = (const float*)d_in[10];
    const float* W1_o1    = (const float*)d_in[11];
    const float* b1_o1    = (const float*)d_in[12];
    const float* W2_o1    = (const float*)d_in[13];
    const float* W1_o2    = (const float*)d_in[14];
    const float* b1_o2    = (const float*)d_in[15];
    const float* W2_o2    = (const float*)d_in[16];
    const int*   ei       = (const int*)d_in[17];

    const int N = in_sizes[3];     // c has N elements
    const int E = in_sizes[2];
    const long N64 = (long)N * 64;

    float* ws    = (float*)d_ws;
    float* scal  = ws;             // [0]=sum(w^2), [1]=rsqrt(sum)
    float* conv  = ws + 64;
    float* conv2 = conv  + N64;
    float* cons  = conv2 + N64;
    float* var   = cons  + N64;
    float* var_k = var   + N64;

    float* out1 = (float*)d_out;
    float* out2 = out1 + N;

    const int BLK = 256;
    const int gemmGrid = ((N / 16) + 7) / 8;            // 8 waves/block, 16x64 per wave
    const long ew4     = N64 / 4;
    const int ewGrid   = (int)((ew4 + BLK - 1) / BLK);
    const long spmmThreads = (long)E * 16;
    const int spmmGrid = (int)((spmmThreads + BLK - 1) / BLK);

    // zero norm scalar + both accumulators (conv & conv2 are contiguous)
    k_zero4<<<1, 64, 0, stream>>>((float4*)scal, 16);
    k_zero4<<<2048, BLK, 0, stream>>>((float4*)conv, 2 * N64 / 4);

    // edge-weight normalization factor
    k_norm_sq<<<512, BLK, 0, stream>>>(edge_w, scal, E);
    k_norm_finalize<<<1, 32, 0, stream>>>(scal);

    // input embeddings (WMMA, K=32)
    k_gemm_bias_relu<<<gemmGrid, BLK, 0, stream>>>(con_feat, W_cons, b_cons, cons, N, 32);
    k_gemm_bias_relu<<<gemmGrid, BLK, 0, stream>>>(var_feat, W_var,  b_var,  var,  N, 32);

    // message-passing layers
    for (int i = 0; i < 4; ++i) {
        k_spmm1<<<spmmGrid, BLK, 0, stream>>>(ei, edge_w, scal, cons, conv, E);
        k_update_var<<<ewGrid, BLK, 0, stream>>>((float4*)var, (float4*)var_k,
                                                 (float4*)conv, c, t_ctov, 2 * i + 1, ew4);
        k_spmm2<<<spmmGrid, BLK, 0, stream>>>(ei, edge_w, scal, var, var_k, conv2, E);
        k_update_cons<<<ewGrid, BLK, 0, stream>>>((float4*)cons, (float4*)conv2,
                                                  b, t_vtoc, 2 * i + 0, ew4);
    }

    // fused output heads (WMMA K=64 + in-register dot with W2)
    k_head<<<gemmGrid, BLK, 0, stream>>>(var,  W1_o1, b1_o1, W2_o1, out1, N);
    k_head<<<gemmGrid, BLK, 0, stream>>>(cons, W1_o2, b1_o2, W2_o2, out2, N);
}
